// Real_Gated_Linear_Recurrent_Unit_29600914604265
// MI455X (gfx1250) — matrix operand order repaired and checked by reading, not compiled, not run on hardware
//
#include <hip/hip_runtime.h>
#include <hip/hip_bf16.h>

typedef __bf16 bf16;
typedef __bf16 v16bf __attribute__((ext_vector_type(16)));
typedef __bf16 v8bf  __attribute__((ext_vector_type(8)));
typedef __bf16 v4bf  __attribute__((ext_vector_type(4)));
typedef float  v8f   __attribute__((ext_vector_type(8)));

namespace {
constexpr int kB  = 4;
constexpr int kS  = 4096;
constexpr int kD  = 1024;
constexpr int kM  = kB * kS;     // 16384 flattened rows
constexpr int kNC = 32;          // scan chunks per sequence
constexpr int kCL = kS / kNC;    // 128 steps per chunk
constexpr float kC = 8.0f;
}

// ---------------------------------------------------------------------------
// fp32 -> bf16 conversion (4 elements / thread, packed stores)
// ---------------------------------------------------------------------------
__global__ __launch_bounds__(256) void rglru_cvt_bf16(
    const float* __restrict__ src, bf16* __restrict__ dst, int n)
{
  int i = (int)(blockIdx.x * blockDim.x + threadIdx.x) << 2;
  if (i < n) {
    float4 f = *(const float4*)(src + i);
    v4bf o;
    o[0] = (bf16)f.x; o[1] = (bf16)f.y; o[2] = (bf16)f.z; o[3] = (bf16)f.w;
    *(v4bf*)(dst + i) = o;
  }
}

// ---------------------------------------------------------------------------
// Fused dual-GEMM (x@Wa^T, x@Wx^T) + gate epilogue.
// Each wave: 32(M) x 16(N) strip, both gates, K looped in steps of 32.
// A (x) and B (W) are both K-contiguous, so A and B share one contiguous k order:
//   A lane l : row mBase+(l&15), 16 contiguous K at 16*(l>>4)
//   B lane l : weight row n, 16 contiguous K at 16*(l>>4)
//   C lane l : VGPR r -> (row r+8*(l>>4), col l&15)
// ---------------------------------------------------------------------------
__global__ __launch_bounds__(256) void rglru_gates_gemm(
    const bf16* __restrict__ Xb, const bf16* __restrict__ Wab,
    const bf16* __restrict__ Wxb, const float* __restrict__ X,
    const float* __restrict__ ba, const float* __restrict__ bx,
    const float* __restrict__ Lam,
    float* __restrict__ aBuf, float* __restrict__ bOut)
{
  const int lane = (int)(threadIdx.x & 31u);
  const int wave = (int)(threadIdx.x >> 5u);
  const int w    = (int)blockIdx.x * 8 + wave;     // 32768 waves total
  const int nTile = w & 63;                        // 64 N-tiles
  const int mBlk  = w >> 6;                        // 512 M-blocks of 32 rows

  const int n     = nTile * 16 + (lane & 15);
  const int mBase = mBlk * 32;
  const int half  = lane >> 4;                     // 0 or 1

  const bf16* pA0 = Xb  + (size_t)(mBase + (lane & 15)) * kD + half * 16;
  const bf16* pA1 = pA0 + (size_t)16 * kD;
  const bf16* pBa = Wab + (size_t)n * kD + half * 16;
  const bf16* pBx = Wxb + (size_t)n * kD + half * 16;

  v8f accA0 = {}; v8f accA1 = {}; v8f accI0 = {}; v8f accI1 = {};

  union AFrag { v16bf v; v8bf h[2]; };

  for (int k = 0; k < kD; k += 32) {
    AFrag a0, a1;
    a0.h[0] = *(const v8bf*)(pA0 + k);
    a0.h[1] = *(const v8bf*)(pA0 + k + 8);
    a1.h[0] = *(const v8bf*)(pA1 + k);
    a1.h[1] = *(const v8bf*)(pA1 + k + 8);
    v16bf fa = *(const v16bf*)(pBa + k);
    v16bf fx = *(const v16bf*)(pBx + k);
    accA0 = __builtin_amdgcn_wmma_f32_16x16x32_bf16(false, a0.v, false, fa,
                                                    (short)0, accA0, false, false);
    accA1 = __builtin_amdgcn_wmma_f32_16x16x32_bf16(false, a1.v, false, fa,
                                                    (short)0, accA1, false, false);
    accI0 = __builtin_amdgcn_wmma_f32_16x16x32_bf16(false, a0.v, false, fx,
                                                    (short)0, accI0, false, false);
    accI1 = __builtin_amdgcn_wmma_f32_16x16x32_bf16(false, a1.v, false, fx,
                                                    (short)0, accI1, false, false);
  }

  // Gate epilogue: rt/it = sigmoid(gemm + bias); a = exp(-C*softplus(-Lam)*rt);
  // b = sqrt(1-a^2) * it * x
  const float bav  = ba[n];
  const float bxv  = bx[n];
  const float mult = -kC * __logf(1.0f + __expf(-Lam[n]));

  #pragma unroll
  for (int t = 0; t < 2; ++t) {
    v8f aA = t ? accA1 : accA0;
    v8f aI = t ? accI1 : accI0;
    #pragma unroll
    for (int r = 0; r < 8; ++r) {
      const int m = mBase + t * 16 + half * 8 + r;
      const float rt = 1.0f / (1.0f + __expf(-(aA[r] + bav)));
      const float it = 1.0f / (1.0f + __expf(-(aI[r] + bxv)));
      const float av = __expf(mult * rt);
      const size_t off = (size_t)m * kD + n;
      const float bv = __fsqrt_rn(fmaxf(1.0f - av * av, 0.0f)) * (it * X[off]);
      aBuf[off] = av;   // a_t
      bOut[off] = bv;   // b_t  (lives in d_out until the scan overwrites it)
    }
  }
}

// ---------------------------------------------------------------------------
// Scan pass 1: per (batch, chunk, channel) local scan with h0 = 0.
// Writes local h into hOut in place of b, and within-chunk cumprod(a) into
// aBuf in place of a. Threads map d-contiguous -> coalesced.
// ---------------------------------------------------------------------------
__global__ __launch_bounds__(256) void rglru_scan_chunk(
    float* __restrict__ aBuf, float* __restrict__ hOut)
{
  const int idx = (int)(blockIdx.x * blockDim.x + threadIdx.x); // [0, kB*kNC*kD)
  const int d = idx & (kD - 1);
  const int c = (idx >> 10) & (kNC - 1);
  const int b = idx >> 15;                                      // kNC*kD = 2^15
  size_t off = ((size_t)(b * kS + c * kCL)) * kD + d;
  float h = 0.0f, p = 1.0f;
  for (int s = 0; s < kCL; ++s) {
    const float av = aBuf[off];
    const float bv = hOut[off];
    h = fmaf(av, h, bv);
    p *= av;
    hOut[off] = h;
    aBuf[off] = p;
    off += (size_t)kD;
  }
}

// ---------------------------------------------------------------------------
// Scan pass 2: serial carry composition across the 32 chunks per channel.
// carry[b,c,d] = h entering chunk c.
// ---------------------------------------------------------------------------
__global__ __launch_bounds__(256) void rglru_scan_carry(
    const float* __restrict__ aBuf, const float* __restrict__ hOut,
    float* __restrict__ carry)
{
  const int idx = (int)(blockIdx.x * blockDim.x + threadIdx.x); // [0, kB*kD)
  const int d = idx & (kD - 1);
  const int b = idx >> 10;
  float cv = 0.0f;
  for (int c = 0; c < kNC; ++c) {
    carry[((size_t)(b * kNC + c)) * kD + d] = cv;
    const size_t last = ((size_t)(b * kS + c * kCL + kCL - 1)) * kD + d;
    cv = fmaf(aBuf[last], cv, hOut[last]);
  }
}

// ---------------------------------------------------------------------------
// Scan pass 3: h_global[t] = h_local[t] + carry_in(chunk) * cumprodA[t]
// ---------------------------------------------------------------------------
__global__ __launch_bounds__(256) void rglru_scan_fix(
    const float* __restrict__ aBuf, const float* __restrict__ carry,
    float* __restrict__ hOut)
{
  const size_t idx = (size_t)blockIdx.x * blockDim.x + threadIdx.x;
  const int d = (int)(idx & (kD - 1));
  const int s = (int)((idx >> 10) & (kS - 1));
  const int b = (int)(idx >> 22);                 // kS*kD = 2^22
  const int c = s / kCL;
  hOut[idx] = fmaf(carry[((size_t)(b * kNC + c)) * kD + d], aBuf[idx], hOut[idx]);
}

// ---------------------------------------------------------------------------
// Launch
// ---------------------------------------------------------------------------
extern "C" void kernel_launch(void* const* d_in, const int* in_sizes, int n_in,
                              void* d_out, int out_size, void* d_ws, size_t ws_size,
                              hipStream_t stream)
{
  const float* x   = (const float*)d_in[0];
  const float* Wa  = (const float*)d_in[1];
  const float* Wx  = (const float*)d_in[2];
  const float* ba  = (const float*)d_in[3];
  const float* bx  = (const float*)d_in[4];
  const float* Lam = (const float*)d_in[5];
  float* out = (float*)d_out;

  char* ws = (char*)d_ws;
  float* aBuf  = (float*)ws;  ws += (size_t)kM * kD * sizeof(float);        // 64 MB
  float* carry = (float*)ws;  ws += (size_t)kB * kNC * kD * sizeof(float);  // 512 KB
  bf16*  Xb    = (bf16*)ws;   ws += (size_t)kM * kD * sizeof(bf16);         // 32 MB
  bf16*  Wab   = (bf16*)ws;   ws += (size_t)kD * kD * sizeof(bf16);         //  2 MB
  bf16*  Wxb   = (bf16*)ws;                                                 //  2 MB

  // 1) bf16 conversion of GEMM operands
  rglru_cvt_bf16<<<(kM * kD) / (256 * 4), 256, 0, stream>>>(x, Xb, kM * kD);
  rglru_cvt_bf16<<<(kD * kD) / (256 * 4), 256, 0, stream>>>(Wa, Wab, kD * kD);
  rglru_cvt_bf16<<<(kD * kD) / (256 * 4), 256, 0, stream>>>(Wx, Wxb, kD * kD);

  // 2) fused dual-GEMM + gate epilogue: a -> aBuf, b -> d_out
  rglru_gates_gemm<<<(kM / 32) * (kD / 16) / 8, 256, 0, stream>>>(
      Xb, Wab, Wxb, x, ba, bx, Lam, aBuf, out);

  // 3) chunked linear-recurrence scan
  rglru_scan_chunk<<<(kB * kNC * kD) / 256, 256, 0, stream>>>(aBuf, out);
  rglru_scan_carry<<<(kB * kD) / 256, 256, 0, stream>>>(aBuf, out, carry);
  rglru_scan_fix<<<((size_t)kM * kD) / 256, 256, 0, stream>>>(aBuf, carry, out);
}
